// MoDAAttention_89618787599008
// MI455X (gfx1250) — matrix operand order repaired and checked by reading, not compiled, not run on hardware
//
#include <hip/hip_runtime.h>

// ---------------------------------------------------------------------------
// MoDA attention for MI455X (gfx1250, wave32, WMMA bf16 16x16x32, f32 acc)
// B=8, N=1024, C=768, HKV=2, G=6, D=64, KV len = 2048 (depth + new)
// Flash attention (64-key tiles, K tile staged via global_load_async_to_lds)
// + ping-pong double-buffered bf16 WMMA GEMMs.
// ---------------------------------------------------------------------------

typedef __attribute__((ext_vector_type(8)))  float          v8f;
typedef __attribute__((ext_vector_type(8)))  __bf16         bf16x8;
typedef __attribute__((ext_vector_type(16))) __bf16         bf16x16;
typedef __attribute__((ext_vector_type(8)))  unsigned short us8;
typedef __attribute__((address_space(3))) unsigned short*   lds_us_t;

#define LOG2E 1.44269504088896340736f

__device__ __forceinline__ unsigned short f2bf(float f) {
  unsigned int u = __float_as_uint(f);
  u += 0x7FFFu + ((u >> 16) & 1u);          // round-to-nearest-even
  return (unsigned short)(u >> 16);
}

// A-operand fragment (16-bit A 16x32): per lane 16 contiguous elems at p
__device__ __forceinline__ bf16x16 load_frag_a(const unsigned short* p) {
  bf16x8 lo = *reinterpret_cast<const bf16x8*>(p);
  bf16x8 hi = *reinterpret_cast<const bf16x8*>(p + 8);
  bf16x16 v;
#pragma unroll
  for (int i = 0; i < 8; ++i) { v[i] = lo[i]; v[i + 8] = hi[i]; }
  return v;
}

// B-operand fragment (16-bit B 32x16): per lane 16 contiguous elems at p
__device__ __forceinline__ bf16x16 load_frag_b(const unsigned short* p) {
  bf16x8 lo = *reinterpret_cast<const bf16x8*>(p);
  bf16x8 hi = *reinterpret_cast<const bf16x8*>(p + 8);
  bf16x16 v;
#pragma unroll
  for (int i = 0; i < 8; ++i) { v[i] = lo[i]; v[i + 8] = hi[i]; }
  return v;
}

__device__ __forceinline__ v8f wmma_bf16(bf16x16 a, bf16x16 b, v8f c) {
  return __builtin_amdgcn_wmma_f32_16x16x32_bf16(false, a, false, b,
                                                 (short)0, c, false, false);
}

// CDNA5 async global->LDS copy (16 bytes per lane), tracked with ASYNCcnt.
__device__ __forceinline__ void async_g2l_b128(unsigned lds_off,
                                               const unsigned short* gp) {
  asm volatile("global_load_async_to_lds_b128 %0, %1, off"
               :: "v"(lds_off), "v"(gp) : "memory");
}
__device__ __forceinline__ void wait_async0() {
  asm volatile("s_wait_asynccnt 0" ::: "memory");
}

// ---------------------------------------------------------------------------
// f32 -> bf16 conversion (grid-stride)
// ---------------------------------------------------------------------------
__global__ void f32_to_bf16_kernel(const float* __restrict__ src,
                                   unsigned short* __restrict__ dst, int n) {
  int i = blockIdx.x * blockDim.x + threadIdx.x;
  int stride = gridDim.x * blockDim.x;
  for (; i < n; i += stride) dst[i] = f2bf(src[i]);
}

// depth_k/depth_v (B,1024,HKV,D) -> first half of k_full/v_full (B,2048,HKV,D)
__global__ void depth_to_bf16_kernel(const float* __restrict__ dk,
                                     const float* __restrict__ dv,
                                     unsigned short* __restrict__ kf,
                                     unsigned short* __restrict__ vf) {
  int i = blockIdx.x * blockDim.x + threadIdx.x;
  if (i >= 8 * 1024 * 2 * 64) return;
  int b = i >> 17;                 // 1024*128 = 2^17 per batch
  int r = i & 131071;
  int d = (b << 18) + r;           // 2048*128 = 2^18 per batch, front half
  kf[d] = f2bf(dk[i]);
  vf[d] = f2bf(dv[i]);
}

// ---------------------------------------------------------------------------
// GEMM:  C[m][n] = sum_k A[m][k] * W[n][k] + bias[n]
// A: (M,K) bf16 row-major.  W: (Nout,K) bf16 row-major (= B operand order).
// Wave = 16x64 strip (4 C tiles).  K-loop ping-pongs two fragment sets by 64.
// mode 0: -> q_bf16.  mode 1: -> k/v f32 outputs + k_full/v_full bf16 tails.
// mode 2: -> f32 final output.
// ---------------------------------------------------------------------------
__global__ __launch_bounds__(256)
void gemm_bf16_kernel(const unsigned short* __restrict__ A,
                      const unsigned short* __restrict__ W,
                      const float* __restrict__ bias,
                      int M, int Nout, int K, int mode,
                      unsigned short* __restrict__ out_bf,
                      float* __restrict__ out_f0,
                      float* __restrict__ out_f1,
                      unsigned short* __restrict__ kf_bf,
                      unsigned short* __restrict__ vf_bf) {
  const int lane = threadIdx.x & 31;
  const int hf   = lane >> 4;
  const int l16  = lane & 15;
  const int wid  = blockIdx.x * (blockDim.x >> 5) + (threadIdx.x >> 5);
  const int ntiles = Nout >> 6;
  const int m0 = (wid / ntiles) << 4;
  const int n0 = (wid % ntiles) << 6;
  if (m0 >= M) return;

  v8f acc[4];
#pragma unroll
  for (int t = 0; t < 4; ++t) acc[t] = (v8f)0.0f;

  const unsigned short* ap = A + (long)(m0 + l16) * K + hf * 16;
  const unsigned short* wp[4];
#pragma unroll
  for (int t = 0; t < 4; ++t)
    wp[t] = W + (long)(n0 + t * 16 + l16) * K + hf * 16;

  // prologue: two fragment sets in flight
  bf16x16 a0 = load_frag_a(ap);
  bf16x16 b0[4];
#pragma unroll
  for (int t = 0; t < 4; ++t) b0[t] = load_frag_b(wp[t]);
  bf16x16 a1 = load_frag_a(ap + 32);
  bf16x16 b1[4];
#pragma unroll
  for (int t = 0; t < 4; ++t) b1[t] = load_frag_b(wp[t] + 32);

  // steady state: consume a set, immediately refill it in place
  for (int kk = 64; kk < K; kk += 64) {
#pragma unroll
    for (int t = 0; t < 4; ++t) acc[t] = wmma_bf16(a0, b0[t], acc[t]);
    a0 = load_frag_a(ap + kk);
#pragma unroll
    for (int t = 0; t < 4; ++t) b0[t] = load_frag_b(wp[t] + kk);
#pragma unroll
    for (int t = 0; t < 4; ++t) acc[t] = wmma_bf16(a1, b1[t], acc[t]);
    a1 = load_frag_a(ap + kk + 32);
#pragma unroll
    for (int t = 0; t < 4; ++t) b1[t] = load_frag_b(wp[t] + kk + 32);
  }
#pragma unroll
  for (int t = 0; t < 4; ++t) acc[t] = wmma_bf16(a0, b0[t], acc[t]);
#pragma unroll
  for (int t = 0; t < 4; ++t) acc[t] = wmma_bf16(a1, b1[t], acc[t]);

#pragma unroll
  for (int t = 0; t < 4; ++t) {
#pragma unroll
    for (int r = 0; r < 8; ++r) {
      const int m = m0 + r + 8 * hf;
      const int n = n0 + t * 16 + l16;
      const float val = acc[t][r] + bias[n];
      if (mode == 0) {
        out_bf[(long)m * Nout + n] = f2bf(val);
      } else if (mode == 2) {
        out_f0[(long)m * Nout + n] = val;
      } else {
        // kv: channel n -> (s,h,d) = (n>>7,(n>>6)&1,n&63); m = b*1024+tok
        const int bb = m >> 10, tok = m & 1023;
        if (n < 128) {
          out_f0[(long)m * 128 + n] = val;
          kf_bf[(long)(bb * 2048 + 1024 + tok) * 128 + n] = f2bf(val);
        } else {
          const int c2 = n - 128;
          out_f1[(long)m * 128 + c2] = val;
          vf_bf[(long)(bb * 2048 + 1024 + tok) * 128 + c2] = f2bf(val);
        }
      }
    }
  }
}

// ---------------------------------------------------------------------------
// Flash attention.  Grid: b(8) x h(2) x qblock(96).  128 threads = 4 waves.
// Wave owns 16 Q rows x D=64.  K loop in 64-key tiles (32 iterations):
//   1. issue async DMA of the shared K tile into LDS (ASYNCcnt)
//   2. issue V tile global loads into registers
//   3. s_wait_asynccnt 0 + barrier  (K tile ready; Vt safe to overwrite)
//   4. 8 S WMMAs (K fragments from LDS -- shared by all 4 waves)
//   5. transposed V store to LDS (b32 key pairs)
//   6. barrier
//   7. online softmax (scaled-log2 domain), P staged to per-wave LDS
//   8. 8 PV WMMAs (V fragments from LDS)
// Q rows map to tokens: row = tok*6 + g, channel = g*128 + h*64 + d.
// ---------------------------------------------------------------------------
__global__ __launch_bounds__(128)
void attn_kernel(const unsigned short* __restrict__ Q,
                 const unsigned short* __restrict__ KF,
                 const unsigned short* __restrict__ VF,
                 unsigned short* __restrict__ O) {
  __shared__ __align__(16) unsigned short Kt[64][72];      // K tile, 64 keys
  __shared__ __align__(16) unsigned short Vt[64][72];      // V^T tile
  __shared__ __align__(16) unsigned short Pst[4][16][72];  // per-wave P tile

  const int lane = threadIdx.x & 31;
  const int hf   = lane >> 4;
  const int l16  = lane & 15;
  const int w    = threadIdx.x >> 5;

  const int qb = blockIdx.x % 96;
  const int h  = (blockIdx.x / 96) & 1;
  const int b  = blockIdx.x / 192;

  const int qrow0 = qb * 64 + w * 16;

  // Q fragments (held in registers across the whole K loop)
  const int mq   = qrow0 + l16;
  const int ntok = mq / 6, g = mq % 6;
  const unsigned short* qrow =
      Q + (long)(b * 1024 + ntok) * 768 + g * 128 + h * 64;
  const bf16x16 aQ0 = load_frag_a(qrow + 0  + hf * 16);
  const bf16x16 aQ1 = load_frag_a(qrow + 32 + hf * 16);

  const float scale2 = 0.125f * LOG2E;     // fold scale and log2(e) together
  float rmax[8], rsum[8];
  v8f accO[4];
#pragma unroll
  for (int r = 0; r < 8; ++r) { rmax[r] = -3.0e38f; rsum[r] = 0.0f; }
#pragma unroll
  for (int t = 0; t < 4; ++t) accO[t] = (v8f)0.0f;

  // cooperative staging roles for this thread
  const int kk0 = (threadIdx.x >> 2) * 2;        // V: key pair 0,2,..,62
  const int d0  = (threadIdx.x & 3) * 16;        // V: d chunk 0,16,32,48
  const int kkk = threadIdx.x >> 1;              // K: key 0..63
  const int kdb = (threadIdx.x & 1) * 32;        // K: d half 0 or 32

  for (int kb = 0; kb < 2048; kb += 64) {
    // --- 1. async DMA of K tile into LDS (64B per thread) ----------------
    {
      const unsigned short* gk =
          KF + (long)((b * 2048 + kb + kkk) * 2 + h) * 64 + kdb;
#pragma unroll
      for (int u = 0; u < 4; ++u) {
        const unsigned loff =
            (unsigned)(unsigned long)(lds_us_t)&Kt[kkk][kdb + u * 8];
        async_g2l_b128(loff, gk + u * 8);
      }
    }

    // --- 2. V tile global loads into registers ---------------------------
    const unsigned short* vp0 =
        VF + (long)((b * 2048 + kb + kk0) * 2 + h) * 64 + d0;
    const unsigned short* vp1 =
        VF + (long)((b * 2048 + kb + kk0 + 1) * 2 + h) * 64 + d0;
    const us8 va0 = *reinterpret_cast<const us8*>(vp0);
    const us8 va1 = *reinterpret_cast<const us8*>(vp0 + 8);
    const us8 vb0 = *reinterpret_cast<const us8*>(vp1);
    const us8 vb1 = *reinterpret_cast<const us8*>(vp1 + 8);

    // prefetch next K/V tile into caches (global_prefetch_b8)
    if (kb + 64 < 2048) {
      const long pofs =
          (long)((b * 2048 + kb + 64 + (threadIdx.x & 63)) * 2 + h) * 64;
      __builtin_prefetch(KF + pofs, 0, 1);
      __builtin_prefetch(VF + pofs, 0, 1);
    }

    // --- 3. K tile ready; prior-iteration Vt readers all arrived ---------
    wait_async0();
    __syncthreads();

    // --- 4. S = Q*K^T: all 8 K fragments from LDS, then 8 WMMAs ----------
    bf16x16 bk[8];
#pragma unroll
    for (int nt = 0; nt < 4; ++nt) {
      const unsigned short* kp = &Kt[nt * 16 + l16][0];
      bk[2 * nt]     = load_frag_b(kp + hf * 16);
      bk[2 * nt + 1] = load_frag_b(kp + 32 + hf * 16);
    }
    v8f s[4];
#pragma unroll
    for (int nt = 0; nt < 4; ++nt) {
      s[nt] = wmma_bf16(aQ0, bk[2 * nt], (v8f)0.0f);
      s[nt] = wmma_bf16(aQ1, bk[2 * nt + 1], s[nt]);
    }

    // --- 5. transposed V store to LDS: key pairs packed into b32 ---------
#pragma unroll
    for (int j = 0; j < 8; ++j) {
      const unsigned int w0 =
          (unsigned int)va0[j] | ((unsigned int)vb0[j] << 16);
      *reinterpret_cast<unsigned int*>(&Vt[d0 + j][kk0]) = w0;
      const unsigned int w1 =
          (unsigned int)va1[j] | ((unsigned int)vb1[j] << 16);
      *reinterpret_cast<unsigned int*>(&Vt[d0 + 8 + j][kk0]) = w1;
    }
    __syncthreads();   // --- 6 ---

    // --- 7. online softmax (row r+8*hf lives in this 16-lane group) ------
    float fac[8];
#pragma unroll
    for (int r = 0; r < 8; ++r) {
      const float a0 = s[0][r] * scale2;
      const float a1 = s[1][r] * scale2;
      const float a2 = s[2][r] * scale2;
      const float a3 = s[3][r] * scale2;
      float mx = fmaxf(fmaxf(a0, a1), fmaxf(a2, a3));
      mx = fmaxf(mx, __shfl_xor(mx, 1));
      mx = fmaxf(mx, __shfl_xor(mx, 2));
      mx = fmaxf(mx, __shfl_xor(mx, 4));
      mx = fmaxf(mx, __shfl_xor(mx, 8));
      const float mnew = fmaxf(rmax[r], mx);
      fac[r] = exp2f(rmax[r] - mnew);
      const float p0 = exp2f(a0 - mnew);
      const float p1 = exp2f(a1 - mnew);
      const float p2 = exp2f(a2 - mnew);
      const float p3 = exp2f(a3 - mnew);
      float ps = (p0 + p1) + (p2 + p3);
      ps += __shfl_xor(ps, 1);
      ps += __shfl_xor(ps, 2);
      ps += __shfl_xor(ps, 4);
      ps += __shfl_xor(ps, 8);
      rsum[r] = rsum[r] * fac[r] + ps;
      rmax[r] = mnew;
      const int ml = r + 8 * hf;
      Pst[w][ml][l16]      = f2bf(p0);
      Pst[w][ml][16 + l16] = f2bf(p1);
      Pst[w][ml][32 + l16] = f2bf(p2);
      Pst[w][ml][48 + l16] = f2bf(p3);
    }
#pragma unroll
    for (int t = 0; t < 4; ++t)
#pragma unroll
      for (int r = 0; r < 8; ++r) accO[t][r] *= fac[r];

    __builtin_amdgcn_wave_barrier();  // LDS is in-order per wave; no reorder

    // --- 8. O += P*V: load all fragments from LDS, then 8 WMMAs ----------
    const bf16x16 aP0 = load_frag_a(&Pst[w][l16][0] + hf * 16);
    const bf16x16 aP1 = load_frag_a(&Pst[w][l16][0] + 32 + hf * 16);
    bf16x16 bv[8];
#pragma unroll
    for (int t = 0; t < 4; ++t) {
      bv[2 * t]     = load_frag_b(&Vt[t * 16 + l16][hf * 16]);
      bv[2 * t + 1] = load_frag_b(&Vt[t * 16 + l16][32 + hf * 16]);
    }
#pragma unroll
    for (int t = 0; t < 4; ++t) {
      accO[t] = wmma_bf16(aP0, bv[2 * t], accO[t]);
      accO[t] = wmma_bf16(aP1, bv[2 * t + 1], accO[t]);
    }
  }

  // --- epilogue: normalize and scatter to o (channel = g*128 + h*64 + d) --
#pragma unroll
  for (int r = 0; r < 8; ++r) {
    const float inv = 1.0f / rsum[r];
    const int mm  = qrow0 + r + 8 * hf;
    const int tok = mm / 6, gg = mm % 6;
    unsigned short* orow =
        O + (long)(b * 1024 + tok) * 768 + gg * 128 + h * 64;
#pragma unroll
    for (int t = 0; t < 4; ++t)
      orow[t * 16 + l16] = f2bf(accO[t][r] * inv);
  }
}

// ---------------------------------------------------------------------------
extern "C" void kernel_launch(void* const* d_in, const int* in_sizes, int n_in,
                              void* d_out, int out_size, void* d_ws,
                              size_t ws_size, hipStream_t stream) {
  (void)in_sizes; (void)n_in; (void)out_size; (void)ws_size;

  const float* x    = (const float*)d_in[0];
  const float* dk   = (const float*)d_in[1];
  const float* dv   = (const float*)d_in[2];
  const float* Wq   = (const float*)d_in[3];
  const float* bq   = (const float*)d_in[4];
  const float* Wkv  = (const float*)d_in[5];
  const float* bkv  = (const float*)d_in[6];
  const float* Wp   = (const float*)d_in[7];
  const float* bp   = (const float*)d_in[8];

  float* out   = (float*)d_out;                 // (8,1024,768)
  float* k_out = out + 8 * 1024 * 768;          // (8,1024,2,64)
  float* v_out = k_out + 8 * 1024 * 2 * 64;     // (8,1024,2,64)

  // workspace layout (bf16 halves), total ~48.9 MB
  unsigned short* ws   = (unsigned short*)d_ws;
  unsigned short* xbf  = ws;                    // 6291456
  unsigned short* wqb  = xbf + 6291456;         //  589824
  unsigned short* wkvb = wqb + 589824;          //  196608
  unsigned short* wpb  = wkvb + 196608;         //  589824
  unsigned short* qbf  = wpb + 589824;          // 6291456
  unsigned short* kf   = qbf + 6291456;         // 2097152 (B,2048,2,64)
  unsigned short* vf   = kf + 2097152;          // 2097152
  unsigned short* obf  = vf + 2097152;          // 6291456

  f32_to_bf16_kernel<<<2048, 256, 0, stream>>>(x, xbf, 6291456);
  f32_to_bf16_kernel<<<512, 256, 0, stream>>>(Wq, wqb, 589824);
  f32_to_bf16_kernel<<<256, 256, 0, stream>>>(Wkv, wkvb, 196608);
  f32_to_bf16_kernel<<<512, 256, 0, stream>>>(Wp, wpb, 589824);
  depth_to_bf16_kernel<<<4096, 256, 0, stream>>>(dk, dv, kf, vf);

  // q = x@Wq.T + bq            (8192 x 768 x 768)  -> q_bf16
  gemm_bf16_kernel<<<768, 256, 0, stream>>>(xbf, wqb, bq, 8192, 768, 768, 0,
                                            qbf, nullptr, nullptr, nullptr,
                                            nullptr);
  // kv = x@Wkv.T + bkv         (8192 x 256 x 768)  -> k,v f32 + kf/vf tails
  gemm_bf16_kernel<<<256, 256, 0, stream>>>(xbf, wkvb, bkv, 8192, 256, 768, 1,
                                            nullptr, k_out, v_out, kf, vf);
  // flash attention            -> o_bf16
  attn_kernel<<<1536, 128, 0, stream>>>(qbf, kf, vf, obf);
  // out = o@Wproj.T + bproj    (8192 x 768 x 768)  -> f32 out
  gemm_bf16_kernel<<<768, 256, 0, stream>>>(obf, wpb, bp, 8192, 768, 768, 2,
                                            nullptr, out, nullptr, nullptr,
                                            nullptr);
}
